// SwinTransformerBlock_1382979470051
// MI455X (gfx1250) — matrix operand order repaired and checked by reading, not compiled, not run on hardware
//
#include <hip/hip_runtime.h>
#include <hip/hip_bf16.h>
#include <math.h>

// ---------------- problem constants ----------------
#define BB   32
#define HH   56
#define WW   56
#define CC   192
#define NHH  6
#define HDD  32          // head dim
#define WSS  7
#define SSS  3
#define NN   49          // tokens per window
#define NWIN 64          // windows per image (8x8)
#define HID  768
#define TT   (BB * HH * WW)   // 100352 tokens

typedef __bf16 v16bf __attribute__((ext_vector_type(16)));
typedef float  v8f   __attribute__((ext_vector_type(8)));

// fp32 -> bf16 round-to-nearest-even
__device__ __forceinline__ __bf16 f2bf(float f) {
  union { float f; unsigned int u; } in; in.f = f;
  unsigned int r = (in.u + 0x7FFFu + ((in.u >> 16) & 1u)) >> 16;
  union { unsigned short s; __bf16 b; } out; out.s = (unsigned short)r;
  return out.b;
}

// ---- WMMA fragment loaders (wave32 16-bit; contiguous k order k=16*(lane/16)+j, same for A and B) ----
// A 16x32 (MxK): lane L holds row M=L&15; lanes<16: K=0..15 ;
//                lanes>=16: K=16..31 (relative to k_base).
__device__ __forceinline__ v16bf load_a_frag(const __bf16* lds, int stride,
                                             int row_base, int k_base) {
  int lane = threadIdx.x & 31;
  const __bf16* p = lds + (row_base + (lane & 15)) * stride + k_base
                        + ((lane & 16) ? 16 : 0);
  v16bf a;
#pragma unroll
  for (int j = 0; j < 16; ++j) a[j] = p[j];
  return a;
}

// B 32x16 (KxN) supplied as B^T in LDS: bt[n][k]. lane holds col N=lane&15;
// lanes<16: K=0..15 ; lanes>=16: K=16..31 (relative to k_base).
__device__ __forceinline__ v16bf load_bt_frag(const __bf16* lds, int stride,
                                              int n_base, int k_base) {
  int lane = threadIdx.x & 31;
  const __bf16* p = lds + (n_base + (lane & 15)) * stride + k_base
                        + ((lane & 16) ? 16 : 0);
  v16bf b;
#pragma unroll
  for (int j = 0; j < 16; ++j) b[j] = p[j];
  return b;
}

__device__ __forceinline__ v8f wmma_bf16(v16bf a, v16bf b, v8f c) {
  return __builtin_amdgcn_wmma_f32_16x16x32_bf16(false, a, false, b,
                                                 (short)0, c, false, false);
}

// ---------------- LayerNorm (wave per token) ----------------
__global__ void __launch_bounds__(256)
ln_kernel(const float* __restrict__ x, const float* __restrict__ g,
          const float* __restrict__ b, __bf16* __restrict__ out) {
  int lane = threadIdx.x & 31;
  int wid  = threadIdx.x >> 5;
  int tok  = blockIdx.x * 8 + wid;
  if (tok >= TT) return;
  const float* row = x + (size_t)tok * CC;
  float vals[6];
  float s = 0.f;
#pragma unroll
  for (int i = 0; i < 6; ++i) { vals[i] = row[i * 32 + lane]; s += vals[i]; }
#pragma unroll
  for (int off = 16; off > 0; off >>= 1) s += __shfl_xor(s, off);
  float mean = s * (1.f / CC);
  float vs = 0.f;
#pragma unroll
  for (int i = 0; i < 6; ++i) { float d = vals[i] - mean; vs += d * d; }
#pragma unroll
  for (int off = 16; off > 0; off >>= 1) vs += __shfl_xor(vs, off);
  float inv = rsqrtf(vs * (1.f / CC) + 1e-5f);
  __bf16* orow = out + (size_t)tok * CC;
#pragma unroll
  for (int i = 0; i < 6; ++i) {
    int c = i * 32 + lane;
    orow[c] = f2bf((vals[i] - mean) * inv * g[c] + b[c]);
  }
}

// ---------------- generic bf16 WMMA GEMM ----------------
// out = A[M,K](bf16) @ W[K,Nw](fp32->bf16) + bias, with templated epilogue.
// EPI 0: qkv  (scale cols<192 by hd^-0.5, write bf16)
// EPI 1: proj (window-reverse + un-shift scatter + residual, write fp32)
// EPI 2: fc1  (exact GELU, write bf16)
// EPI 3: fc2  (accumulate into fp32 out)
template <int EPI>
__global__ void __launch_bounds__(256)
gemm_bf16(const __bf16* __restrict__ A, const float* __restrict__ W,
          const float* __restrict__ bias, int K, int Nw,
          __bf16* __restrict__ out_bf, float* __restrict__ out_f,
          const float* __restrict__ resid) {
  __shared__ __bf16 As[64][40];   // [m][k]
  __shared__ __bf16 Bt[64][40];   // [n][k]
  int tid = threadIdx.x;
  int lane = tid & 31, wid = tid >> 5;
  int m0 = blockIdx.x * 64, n0 = blockIdx.y * 64;
  int wm = wid & 3;        // row tile (16 rows)
  int wn = wid >> 2;       // col group (32 cols)
  v8f acc0 = {}; v8f acc1 = {};

  for (int kk = 0; kk < K; kk += 32) {
    { // stage A tile 64x32 (bf16, 16B per thread)
      int row = tid >> 2, ch = tid & 3;
      const uint4* src = (const uint4*)(A + (size_t)(m0 + row) * K + kk);
      *(uint4*)&As[row][ch * 8] = src[ch];
    }
    { // stage W tile 32x64 (fp32 -> bf16), transposed into Bt[n][k]
#pragma unroll
      for (int r = 0; r < 2; ++r) {
        int c = tid + r * 256;        // 0..511 float4 chunks
        int k = c >> 4, nq = c & 15;
        float4 wv = *(const float4*)(W + (size_t)(kk + k) * Nw + n0 + nq * 4);
        Bt[nq * 4 + 0][k] = f2bf(wv.x);
        Bt[nq * 4 + 1][k] = f2bf(wv.y);
        Bt[nq * 4 + 2][k] = f2bf(wv.z);
        Bt[nq * 4 + 3][k] = f2bf(wv.w);
      }
    }
    __syncthreads();
    v16bf a  = load_a_frag(&As[0][0], 40, wm * 16, 0);
    v16bf b0 = load_bt_frag(&Bt[0][0], 40, wn * 32, 0);
    v16bf b1 = load_bt_frag(&Bt[0][0], 40, wn * 32 + 16, 0);
    acc0 = wmma_bf16(a, b0, acc0);
    acc1 = wmma_bf16(a, b1, acc1);
    __syncthreads();
  }

  // epilogue: lane holds col n=lane&15, rows v + (lane<16?0:8)
  int half = (lane & 16) ? 8 : 0;
  int nl = lane & 15;
#pragma unroll
  for (int t = 0; t < 2; ++t) {
    int gc = n0 + wn * 32 + t * 16 + nl;
    float bval = bias[gc];
#pragma unroll
    for (int v = 0; v < 8; ++v) {
      int gr = m0 + wm * 16 + v + half;
      float val = (t == 0 ? acc0[v] : acc1[v]) + bval;
      if (EPI == 0) {                 // qkv: scale q columns
        if (gc < CC) val *= 0.17677669529663687f;   // 1/sqrt(32)
        out_bf[(size_t)gr * Nw + gc] = f2bf(val);
      } else if (EPI == 1) {          // proj: scatter + residual
        int w = gr / NN, pos = gr % NN;
        int bi = w >> 6, wy = (w >> 3) & 7, wx = w & 7;
        int p = pos / WSS, q = pos % WSS;
        int h = wy * WSS + p + SSS; if (h >= HH) h -= HH;
        int c = wx * WSS + q + SSS; if (c >= WW) c -= WW;
        size_t o = ((size_t)bi * (HH * WW) + h * WW + c) * CC + gc;
        out_f[o] = resid[o] + val;
      } else if (EPI == 2) {          // fc1: exact GELU
        val = 0.5f * val * (1.f + erff(val * 0.7071067811865476f));
        out_bf[(size_t)gr * Nw + gc] = f2bf(val);
      } else {                        // fc2: residual accumulate
        size_t o = (size_t)gr * CC + gc;
        out_f[o] += val;
      }
    }
  }
}

// ---------------- windowed attention (one wave per window*head) ----------------
__global__ void __launch_bounds__(32)
attn_kernel(const __bf16* __restrict__ qkv, const float* __restrict__ rel_bias,
            __bf16* __restrict__ xw) {
  int w = blockIdx.x;            // 0..2047 : b*64 + wy*8 + wx
  int head = blockIdx.y;         // 0..5
  int bi = w >> 6, wy = (w >> 3) & 7, wx = w & 7;
  int lane = threadIdx.x;

  __shared__ __bf16 Q[64][40];    // [query][d]   rows>=49 zero
  __shared__ __bf16 Kl[64][40];   // [key][d]   == B^T for QK^T
  __shared__ __bf16 Vt[32][72];   // [d][key]   == B^T for A*V
  __shared__ __bf16 P[64][72];    // probs (unnormalized) [query][key]
  __shared__ float  S[64][65];    // scores
  __shared__ float  rsum[64];

  // gather q/k/v for this window+head with the cyclic shift applied
#pragma unroll
  for (int rep = 0; rep < 2; ++rep) {
    int i = lane + rep * 32;
    if (i < NN) {
      int p = i / WSS, q = i % WSS;
      int h = wy * WSS + p + SSS; if (h >= HH) h -= HH;
      int c = wx * WSS + q + SSS; if (c >= WW) c -= WW;
      size_t tok = (size_t)bi * (HH * WW) + h * WW + c;
      const __bf16* base = qkv + tok * (3 * CC) + head * HDD;
      const uint4* qs = (const uint4*)base;
      const uint4* ks = (const uint4*)(base + CC);
      uint4* qd = (uint4*)&Q[i][0];
      uint4* kd = (uint4*)&Kl[i][0];
#pragma unroll
      for (int t = 0; t < 4; ++t) { qd[t] = qs[t]; kd[t] = ks[t]; }
      const __bf16* vs = base + 2 * CC;
#pragma unroll
      for (int d = 0; d < HDD; ++d) Vt[d][i] = vs[d];
    } else {
      uint4 z = make_uint4(0, 0, 0, 0);
      uint4* qd = (uint4*)&Q[i][0];
      uint4* kd = (uint4*)&Kl[i][0];
#pragma unroll
      for (int t = 0; t < 4; ++t) { qd[t] = z; kd[t] = z; }
      __bf16 zb = f2bf(0.f);
#pragma unroll
      for (int d = 0; d < HDD; ++d) Vt[d][i] = zb;
    }
  }
  __syncthreads();

  int half = (lane & 16) ? 8 : 0;
  int nl = lane & 15;

  // ---- S = (q * hd^-0.5) @ k^T  (scale already folded into q) ----
  v16bf aq[4];
#pragma unroll
  for (int tr = 0; tr < 4; ++tr) aq[tr] = load_a_frag(&Q[0][0], 40, tr * 16, 0);
#pragma unroll
  for (int tc = 0; tc < 4; ++tc) {
    v16bf bk = load_bt_frag(&Kl[0][0], 40, tc * 16, 0);
#pragma unroll
    for (int tr = 0; tr < 4; ++tr) {
      v8f acc = {};
      acc = wmma_bf16(aq[tr], bk, acc);
#pragma unroll
      for (int v = 0; v < 8; ++v) S[tr * 16 + v + half][tc * 16 + nl] = acc[v];
    }
  }
  __syncthreads();

  // ---- softmax with relative-position bias + shift mask ----
#pragma unroll
  for (int rep = 0; rep < 2; ++rep) {
    int r = lane + rep * 32;
    if (r < NN) {
      int ih = r / WSS, iw = r % WSS;
      int hi = wy * WSS + ih, wi = wx * WSS + iw;   // shifted-image coords
      int ri = (hi < 49 ? 0 : (hi < 53 ? 1 : 2)) * 3
             + (wi < 49 ? 0 : (wi < 53 ? 1 : 2));
      float mx = -1e30f;
      for (int j = 0; j < NN; ++j) {
        int jh = j / WSS, jw = j % WSS;
        int hj = wy * WSS + jh, wj = wx * WSS + jw;
        int rj = (hj < 49 ? 0 : (hj < 53 ? 1 : 2)) * 3
               + (wj < 49 ? 0 : (wj < 53 ? 1 : 2));
        int idx = (ih - jh + WSS - 1) * (2 * WSS - 1) + (iw - jw + WSS - 1);
        float val = S[r][j] + rel_bias[idx * NHH + head]
                  + ((ri != rj) ? -100.f : 0.f);
        S[r][j] = val;
        mx = fmaxf(mx, val);
      }
      float sum = 0.f;
      for (int j = 0; j < NN; ++j) {
        float e = __expf(S[r][j] - mx);
        sum += e;
        P[r][j] = f2bf(e);
      }
      for (int j = NN; j < 64; ++j) P[r][j] = f2bf(0.f);
      rsum[r] = sum;
    } else {
      for (int j = 0; j < 64; ++j) P[r][j] = f2bf(0.f);
      rsum[r] = 1.f;
    }
  }
  __syncthreads();

  // ---- out = softmax(S) @ v  (normalize by rsum in epilogue) ----
#pragma unroll
  for (int tr = 0; tr < 4; ++tr) {
    v16bf a0 = load_a_frag(&P[0][0], 72, tr * 16, 0);
    v16bf a1 = load_a_frag(&P[0][0], 72, tr * 16, 32);
#pragma unroll
    for (int tc = 0; tc < 2; ++tc) {
      v16bf b0 = load_bt_frag(&Vt[0][0], 72, tc * 16, 0);
      v16bf b1 = load_bt_frag(&Vt[0][0], 72, tc * 16, 32);
      v8f acc = {};
      acc = wmma_bf16(a0, b0, acc);
      acc = wmma_bf16(a1, b1, acc);
#pragma unroll
      for (int v = 0; v < 8; ++v) {
        int r = tr * 16 + v + half;
        if (r < NN) {
          float o = acc[v] / rsum[r];
          xw[((size_t)w * NN + r) * CC + head * HDD + tc * 16 + nl] = f2bf(o);
        }
      }
    }
  }
}

// ---------------- host launcher ----------------
extern "C" void kernel_launch(void* const* d_in, const int* in_sizes, int n_in,
                              void* d_out, int out_size, void* d_ws, size_t ws_size,
                              hipStream_t stream) {
  const float* x    = (const float*)d_in[0];
  const float* n1g  = (const float*)d_in[1];
  const float* n1b  = (const float*)d_in[2];
  const float* qkvw = (const float*)d_in[3];
  const float* qkvb = (const float*)d_in[4];
  const float* relb = (const float*)d_in[5];
  const float* pw   = (const float*)d_in[6];
  const float* pb   = (const float*)d_in[7];
  const float* n2g  = (const float*)d_in[8];
  const float* n2b  = (const float*)d_in[9];
  const float* f1w  = (const float*)d_in[10];
  const float* f1b  = (const float*)d_in[11];
  const float* f2w  = (const float*)d_in[12];
  const float* f2b  = (const float*)d_in[13];
  float* out = (float*)d_out;

  // workspace: buf0 (T*192 bf16) serves xn -> xw -> xm; buf1 (T*768 bf16) qkv -> h
  char* ws = (char*)d_ws;
  const size_t B0 = (size_t)TT * CC * 2;   // 38,535,168 B
  __bf16* buf0 = (__bf16*)ws;
  __bf16* buf1 = (__bf16*)(ws + B0);

  // 1) LN1 -> xn (bf16)
  ln_kernel<<<TT / 8, 256, 0, stream>>>(x, n1g, n1b, buf0);
  // 2) qkv = xn @ qkv_w + qkv_b   (q pre-scaled by hd^-0.5)
  gemm_bf16<0><<<dim3(TT / 64, (3 * CC) / 64), 256, 0, stream>>>(
      buf0, qkvw, qkvb, CC, 3 * CC, buf1, nullptr, nullptr);
  // 3) windowed attention -> xw (bf16, window order)
  attn_kernel<<<dim3(BB * NWIN, NHH), 32, 0, stream>>>(buf1, relb, buf0);
  // 4) x2 = x + scatter(xw @ proj_w + proj_b)  -> d_out (fp32)
  gemm_bf16<1><<<dim3(TT / 64, CC / 64), 256, 0, stream>>>(
      buf0, pw, pb, CC, CC, nullptr, out, x);
  // 5) LN2 -> xm (bf16)
  ln_kernel<<<TT / 8, 256, 0, stream>>>(out, n2g, n2b, buf0);
  // 6) h = gelu(xm @ fc1_w + fc1_b)  (bf16)
  gemm_bf16<2><<<dim3(TT / 64, HID / 64), 256, 0, stream>>>(
      buf0, f1w, f1b, CC, HID, buf1, nullptr, nullptr);
  // 7) d_out += h @ fc2_w + fc2_b
  gemm_bf16<3><<<dim3(TT / 64, CC / 64), 256, 0, stream>>>(
      buf1, f2w, f2b, HID, CC, nullptr, out, nullptr);
}
